// Cross_Attention_Layer_73598559584454
// MI455X (gfx1250) — matrix operand order repaired and checked by reading, not compiled, not run on hardware
//
#include <hip/hip_runtime.h>
#include <math.h>

typedef __attribute__((ext_vector_type(8)))  _Float16 v8h;
typedef __attribute__((ext_vector_type(16))) _Float16 v16h;
typedef __attribute__((ext_vector_type(8)))  float    v8f;

#define S_ROWS 8192
#define T_ROWS 8192
#define DMODEL 768
#define DK 128
#define DV 64
#define NSPLIT 4
#define TSPLIT (T_ROWS / NSPLIT)  // 2048 keys per split

static __device__ __forceinline__ v8f wmma_f16(v16h a, v16h b, v8f c) {
  // D = A(16x32 f16) * B(32x16 f16) + C(16x16 f32)
  return __builtin_amdgcn_wmma_f32_16x16x32_f16(false, a, false, b, (short)0, c,
                                                false, false);
}

// CDNA5 async DMA: 16 bytes/lane global -> LDS, tracked by ASYNCcnt.
static __device__ __forceinline__ void async_copy16(void* lds,
                                                    const _Float16* g) {
  uint32_t lds_off = (uint32_t)(uintptr_t)lds;  // low 32 bits = LDS byte addr
  asm volatile("global_load_async_to_lds_b128 %0, %1, off" ::"v"(lds_off),
               "v"(g)
               : "memory");
}
static __device__ __forceinline__ void wait_async0() {
  asm volatile("s_wait_asynccnt 0x0" ::: "memory");
}

// ---------------------------------------------------------------------------
// Projection GEMM: out[16x16 tile] = X[M,768] @ W[N,768]^T, f32 in, f16 out.
// One wave per 16x16 output tile; K loop of 24 WMMA(16x16x32) steps.
// transposed==1 stores out[n][m] (used for V so P@V B-operand is contiguous).
// ---------------------------------------------------------------------------
__global__ __launch_bounds__(128) void proj_kernel(
    const float* __restrict__ X, const float* __restrict__ W,
    _Float16* __restrict__ out, int ntiles, int ldOut, int transposed) {
  const int lane = threadIdx.x & 31;
  const int wave = threadIdx.x >> 5;
  const int tile = blockIdx.x * 4 + wave;
  const int m0 = (tile / ntiles) * 16;
  const int n0 = (tile % ntiles) * 16;
  const int col = lane & 15;
  const int hi = lane >> 4;

  v8f acc;
#pragma unroll
  for (int i = 0; i < 8; i++) acc[i] = 0.0f;

  // A layout: lane<16 -> row=lane, K 0..15; lane>=16 -> K 16..31
  const float* xrow = X + (size_t)(m0 + col) * DMODEL + hi * 16;
  // B layout: lane<16 -> col=lane, K 0..15; lane>=16 -> K 16..31
  const float* wrow = W + (size_t)(n0 + col) * DMODEL + hi * 16;

#pragma unroll 4
  for (int k0 = 0; k0 < DMODEL; k0 += 32) {
    v8f xlo = *(const v8f*)(xrow + k0);
    v8f xhi = *(const v8f*)(xrow + k0 + 8);
    v8f w0 = *(const v8f*)(wrow + k0);
    v8f w1 = *(const v8f*)(wrow + k0 + 8);
    v16h a, b;
#pragma unroll
    for (int i = 0; i < 8; i++) {
      a[i] = (_Float16)xlo[i];
      a[i + 8] = (_Float16)xhi[i];
      b[i] = (_Float16)w0[i];
      b[i + 8] = (_Float16)w1[i];
    }
    acc = wmma_f16(a, b, acc);
  }

  // C layout: VGPR r, lane -> row = r + 8*hi, col = lane&15
  if (!transposed) {
#pragma unroll
    for (int r = 0; r < 8; r++)
      out[(size_t)(m0 + r + hi * 8) * ldOut + n0 + col] = (_Float16)acc[r];
  } else {
#pragma unroll
    for (int r = 0; r < 8; r++)
      out[(size_t)(n0 + col) * S_ROWS + m0 + r + hi * 8] = (_Float16)acc[r];
  }
}

// ---------------------------------------------------------------------------
// Flash attention over one key split. One wave owns 16 query rows; block of
// 4 waves shares double-buffered K/V tiles staged via async LDS DMA.
// Emits unnormalized partials per (row, split): m, l, g = O.w_out.
// ---------------------------------------------------------------------------
__global__ __launch_bounds__(128) void attn_kernel(
    const _Float16* __restrict__ Q, const _Float16* __restrict__ K,
    const _Float16* __restrict__ Vt, const float* __restrict__ w_out,
    float* __restrict__ mpart, float* __restrict__ lpart,
    float* __restrict__ gpart) {
  const int tid = threadIdx.x;
  const int lane = tid & 31;
  const int wave = tid >> 5;
  const int qr0 = (blockIdx.x * 4 + wave) * 16;
  const int split = blockIdx.y;
  const int t0base = split * TSPLIT;
  const int col = lane & 15;
  const int hi = lane >> 4;

  __shared__ _Float16 kbuf[2][32 * DK];  // 8 KB each
  __shared__ _Float16 vbuf[2][DV * 32];  // 4 KB each
  __shared__ _Float16 sP[4][16 * 32];    // per-wave C->A re-layout staging
  _Float16* sp = sP[wave];

  // Preload all 4 Q A-tiles (full d_k=128) for this wave's 16 rows.
  v16h aQ[4];
  {
    const _Float16* qrow = Q + (size_t)(qr0 + col) * DK + hi * 16;
#pragma unroll
    for (int kk = 0; kk < 4; kk++) {
      v8h lo = *(const v8h*)(qrow + kk * 32);
      v8h hh = *(const v8h*)(qrow + kk * 32 + 8);
#pragma unroll
      for (int i = 0; i < 8; i++) {
        aQ[kk][i] = lo[i];
        aQ[kk][i + 8] = hh[i];
      }
    }
  }
  float wv[4];
#pragma unroll
  for (int j = 0; j < 4; j++) wv[j] = w_out[j * 16 + col];

  float mrow[8], lrow[8];
  v8f accO[4];
#pragma unroll
  for (int r = 0; r < 8; r++) {
    mrow[r] = -INFINITY;
    lrow[r] = 0.0f;
  }
#pragma unroll
  for (int j = 0; j < 4; j++)
#pragma unroll
    for (int i = 0; i < 8; i++) accO[j][i] = 0.0f;

  const float inv_sqrt = 0.088388347648318447f;  // 1/sqrt(128)
  const float log2e = 1.4426950408889634f;
  const int nT = TSPLIT / 32;

  // Stage a 32-key K/V tile into buffer b. 6 async b128 ops per lane.
  auto stage = [&](int b, int t0) {
    const _Float16* kg = K + (size_t)t0 * DK;  // 32 rows contiguous: 8 KB flat
#pragma unroll
    for (int r = 0; r < 4; ++r) {
      int c = tid + r * 128;  // 512 16-byte chunks
      async_copy16(&kbuf[b][c * 8], kg + c * 8);
    }
#pragma unroll
    for (int r = 0; r < 2; ++r) {
      int c = tid + r * 128;  // 256 16-byte chunks: 64 rows x 32 halfs
      int row = c >> 2, off = (c & 3) * 8;
      async_copy16(&vbuf[b][row * 32 + off],
                   Vt + (size_t)row * T_ROWS + t0 + off);
    }
  };

  stage(0, t0base);  // prologue

  for (int it = 0; it < nT; ++it) {
    const int cur = it & 1;
    const int t0 = t0base + it * 32;
    wait_async0();    // our share of tile `it` is in LDS
    __syncthreads();  // everyone's share is in; prev tile fully consumed
    {
      const int tn = (it + 1 < nT) ? it + 1 : 0;  // harmless wrap prefetch
      stage(cur ^ 1, t0base + tn * 32);
    }
    const _Float16* kb = kbuf[cur];
    const _Float16* vb = vbuf[cur];
    (void)t0;

    // ---- scores for two 16-key tiles ----
    v8f sacc[2];
#pragma unroll
    for (int ntv = 0; ntv < 2; ntv++) {
      v8f acc;
#pragma unroll
      for (int i = 0; i < 8; i++) acc[i] = 0.0f;
      const _Float16* krow = kb + (ntv * 16 + col) * DK + hi * 16;
#pragma unroll
      for (int kk = 0; kk < 4; kk++) {
        v16h b = *(const v16h*)(krow + kk * 32);
        acc = wmma_f16(aQ[kk], b, acc);
      }
#pragma unroll
      for (int i = 0; i < 8; i++) sacc[ntv][i] = acc[i] * inv_sqrt;
    }

    // ---- online softmax (row reductions across 16-lane halves) ----
#pragma unroll
    for (int r = 0; r < 8; r++) {
      float t = fmaxf(sacc[0][r], sacc[1][r]);
      t = fmaxf(t, __shfl_xor(t, 1, 16));
      t = fmaxf(t, __shfl_xor(t, 2, 16));
      t = fmaxf(t, __shfl_xor(t, 4, 16));
      t = fmaxf(t, __shfl_xor(t, 8, 16));
      float mn = fmaxf(mrow[r], t);
      float alpha = exp2f((mrow[r] - mn) * log2e);
      mrow[r] = mn;
      float p0 = exp2f((sacc[0][r] - mn) * log2e);
      float p1 = exp2f((sacc[1][r] - mn) * log2e);
      float ps = p0 + p1;
      ps += __shfl_xor(ps, 1, 16);
      ps += __shfl_xor(ps, 2, 16);
      ps += __shfl_xor(ps, 4, 16);
      ps += __shfl_xor(ps, 8, 16);
      lrow[r] = lrow[r] * alpha + ps;
#pragma unroll
      for (int j = 0; j < 4; j++) accO[j][r] *= alpha;
      const int row = r + hi * 8;
      sp[row * 32 + col] = (_Float16)p0;
      sp[row * 32 + col + 16] = (_Float16)p1;
    }
    __builtin_amdgcn_wave_barrier();
    asm volatile("s_wait_dscnt 0" ::: "memory");

    // ---- reload P in A-matrix layout ----
    v16h aP;
    {
      const _Float16* prow = sp + col * 32 + hi * 16;
      v8h lo = *(const v8h*)(prow);
      v8h hh = *(const v8h*)(prow + 8);
#pragma unroll
      for (int i = 0; i < 8; i++) {
        aP[i] = lo[i];
        aP[i + 8] = hh[i];
      }
    }
    __builtin_amdgcn_wave_barrier();

    // ---- O += P @ V (d_v = 64 -> 4 N-tiles) ----
#pragma unroll
    for (int j = 0; j < 4; j++) {
      v16h bv = *(const v16h*)(vb + (j * 16 + col) * 32 + hi * 16);
      accO[j] = wmma_f16(aP, bv, accO[j]);
    }
  }

  // ---- partials: m, l, g = O . w_out (unnormalized) ----
#pragma unroll
  for (int r = 0; r < 8; r++) {
    float t = accO[0][r] * wv[0] + accO[1][r] * wv[1] + accO[2][r] * wv[2] +
              accO[3][r] * wv[3];
    t += __shfl_xor(t, 1, 16);
    t += __shfl_xor(t, 2, 16);
    t += __shfl_xor(t, 4, 16);
    t += __shfl_xor(t, 8, 16);
    if (col == 0) {
      const int row = qr0 + hi * 8 + r;
      mpart[split * S_ROWS + row] = mrow[r];
      lpart[split * S_ROWS + row] = lrow[r];
      gpart[split * S_ROWS + row] = t;
    }
  }
}

// ---------------------------------------------------------------------------
// Combine key-split partials: gate[s] = sum(e^(m_k-M) g_k) / sum(e^(m_k-M) l_k)
// ---------------------------------------------------------------------------
__global__ __launch_bounds__(256) void combine_kernel(
    const float* __restrict__ mpart, const float* __restrict__ lpart,
    const float* __restrict__ gpart, float* __restrict__ gate) {
  int s = blockIdx.x * blockDim.x + threadIdx.x;
  if (s >= S_ROWS) return;
  const float log2e = 1.4426950408889634f;
  float M = -INFINITY;
#pragma unroll
  for (int k = 0; k < NSPLIT; k++) M = fmaxf(M, mpart[k * S_ROWS + s]);
  float L = 0.0f, G = 0.0f;
#pragma unroll
  for (int k = 0; k < NSPLIT; k++) {
    float e = exp2f((mpart[k * S_ROWS + s] - M) * log2e);
    L += e * lpart[k * S_ROWS + s];
    G += e * gpart[k * S_ROWS + s];
  }
  gate[s] = G / L;
}

// ---------------------------------------------------------------------------
// sum(x2) reduction for the passthrough branch flag.
// ---------------------------------------------------------------------------
__global__ __launch_bounds__(256) void sum_kernel(const float* __restrict__ x,
                                                  float* __restrict__ sum,
                                                  int n) {
  __shared__ float red[256];
  float acc = 0.0f;
  for (int i = blockIdx.x * blockDim.x + threadIdx.x; i < n;
       i += gridDim.x * blockDim.x)
    acc += x[i];
  red[threadIdx.x] = acc;
  __syncthreads();
  for (int s = 128; s > 0; s >>= 1) {
    if ((int)threadIdx.x < s) red[threadIdx.x] += red[threadIdx.x + s];
    __syncthreads();
  }
  if (threadIdx.x == 0) atomicAdd(sum, red[0]);
}

// ---------------------------------------------------------------------------
// out[s,c] = x1[s,c] * (1 - gate[s]); passthrough if sum(x2)==0.
// ---------------------------------------------------------------------------
__global__ __launch_bounds__(256) void gate_kernel(
    const float* __restrict__ x1, const float* __restrict__ gate,
    const float* __restrict__ sumx2, float* __restrict__ out, int total4) {
  int i = blockIdx.x * blockDim.x + threadIdx.x;
  if (i >= total4) return;
  float4 v = ((const float4*)x1)[i];
  int s = i / (DMODEL / 4);
  float g = (sumx2[0] == 0.0f) ? 0.0f : gate[s];
  float sc = 1.0f - g;
  v.x *= sc;
  v.y *= sc;
  v.z *= sc;
  v.w *= sc;
  ((float4*)out)[i] = v;
}

extern "C" void kernel_launch(void* const* d_in, const int* in_sizes, int n_in,
                              void* d_out, int out_size, void* d_ws,
                              size_t ws_size, hipStream_t stream) {
  const float* x1 = (const float*)d_in[0];   // [1,8192,768]
  const float* x2 = (const float*)d_in[1];   // [1,8192,768]
  const float* w_q = (const float*)d_in[2];  // [128,768]
  const float* w_k = (const float*)d_in[3];  // [128,768]
  const float* w_v = (const float*)d_in[4];  // [64,768]
  const float* w_o = (const float*)d_in[5];  // [1,64]
  float* out = (float*)d_out;

  // Workspace layout (~5.7 MB)
  _Float16* Qf = (_Float16*)d_ws;
  _Float16* Kf = Qf + (size_t)S_ROWS * DK;
  _Float16* Vt = Kf + (size_t)T_ROWS * DK;
  float* gate = (float*)(Vt + (size_t)DV * T_ROWS);
  float* sum = gate + S_ROWS;
  float* mpart = sum + 64;  // padded
  float* lpart = mpart + NSPLIT * S_ROWS;
  float* gpart = lpart + NSPLIT * S_ROWS;

  hipMemsetAsync(sum, 0, sizeof(float), stream);
  sum_kernel<<<256, 256, 0, stream>>>(x2, sum, T_ROWS * DMODEL);

  // Q = x1 @ w_q^T ; K = x2 @ w_k^T ; Vt = (x2 @ w_v^T)^T
  proj_kernel<<<(512 * 8) / 4, 128, 0, stream>>>(x1, w_q, Qf, 8, DK, 0);
  proj_kernel<<<(512 * 8) / 4, 128, 0, stream>>>(x2, w_k, Kf, 8, DK, 0);
  proj_kernel<<<(512 * 4) / 4, 128, 0, stream>>>(x2, w_v, Vt, 4, 0, 1);

  attn_kernel<<<dim3(512 / 4, NSPLIT), 128, 0, stream>>>(Qf, Kf, Vt, w_o,
                                                         mpart, lpart, gpart);
  combine_kernel<<<(S_ROWS + 255) / 256, 256, 0, stream>>>(mpart, lpart, gpart,
                                                           gate);

  gate_kernel<<<(S_ROWS * DMODEL / 4 + 255) / 256, 256, 0, stream>>>(
      x1, gate, sum, out, S_ROWS * DMODEL / 4);
}
